// MCPropNetMSHR_85023172591739
// MI455X (gfx1250) — matrix operand order repaired and checked by reading, not compiled, not run on hardware
//
#include <hip/hip_runtime.h>
#include <math.h>

// ---------------------------------------------------------------------------
// MCPropNet-MSHR for MI455X (gfx1250): conv stack via v_wmma_f32_16x16x32_bf16
// with LDS-staged 16x4 pixel tiles (18x6 halo), channels-last bf16, f32 acc.
// ---------------------------------------------------------------------------

typedef __bf16 v16bf __attribute__((ext_vector_type(16)));
typedef float  v8f   __attribute__((ext_vector_type(8)));

struct alignas(16) Q4 { unsigned int a, b, c, d; };   // 16-byte chunk (b128)
union Frag { Q4 q[2]; v16bf v; };                      // 32 bytes = 16 bf16

// ---- bf16 helpers (raw ushort storage, RNE convert) ------------------------
__device__ __forceinline__ unsigned short f2bf(float f) {
  union { float f; unsigned int u; } v; v.f = f;
  unsigned int u = v.u;
  u += 0x7fffu + ((u >> 16) & 1u);
  return (unsigned short)(u >> 16);
}
__device__ __forceinline__ float bf2f(unsigned short h) {
  union { unsigned int u; float f; } v; v.u = ((unsigned int)h) << 16;
  return v.f;
}

// ===========================================================================
// Weight packer: OIHW f32 -> WMMA B-fragment layout.
// K ordering: K = tap*CpadIn + ci, chunked by 32 (nested: tap outer, ci-block inner).
// Fragment layout (contiguous k per lane half, same order as the A loads):
//   lane l: n = l&15 (output col), khalf = l>>4; dword v holds K = khalf*16+2v (+1).
// Flat: Wp[((ntile*nK + kc)*32 + lane)*8 + v]
// ===========================================================================
__global__ void pack_weights_kernel(const float* __restrict__ w,  // [Cout][CinLog][3][3]
                                    unsigned int* __restrict__ Wp,
                                    int CinLog, int CinUse, int CpadIn,
                                    int Cout, int ntiles) {
  const int cblocks = CpadIn >> 5;
  const int nK = 9 * cblocks;
  const int total = ntiles * nK * 32;
  int idx = blockIdx.x * blockDim.x + threadIdx.x;
  if (idx >= total) return;
  const int lane = idx & 31;
  const int kc = (idx >> 5) % nK;
  const int nt = (idx >> 5) / nK;
  const int n = lane & 15, khalf = lane >> 4;
  const int co = nt * 16 + n;
  const int tap = kc / cblocks, cb = kc % cblocks;
  const int ky = tap / 3, kx = tap % 3;
  unsigned int* dst = Wp + (size_t)idx * 8;
#pragma unroll
  for (int v = 0; v < 8; ++v) {
    unsigned short lo = 0, hi = 0;
    const int k0 = cb * 32 + khalf * 16 + 2 * v;
#pragma unroll
    for (int t = 0; t < 2; ++t) {
      const int ci = k0 + t;
      unsigned short val = 0;
      if (co < Cout && ci < CinUse) {
        float f = w[(((size_t)co * CinLog + ci) * 3 + ky) * 3 + kx];
        val = f2bf(f);
      }
      if (t == 0) lo = val; else hi = val;
    }
    dst[v] = ((unsigned int)hi << 16) | (unsigned int)lo;
  }
}

// ===========================================================================
// LDS-staged implicit-GEMM 3x3 conv via WMMA bf16.
// Block = 128 threads = 4 waves; wave w owns output-channel tile [w*16, w*16+16).
// Tile: 16x4 output pixels; 18x6 zero-padded halo staged in LDS once.
// Inner loop (fully unrolled taps): 2 b128 weight loads reused across 4 M-tiles;
// per tile 2 ds_load_b128 + 1 v_wmma_f32_16x16x32_bf16. No predication inside.
// A fragment per lane: row m = lane&15, ksel = lane>>4; K in
//   [ksel*16, +16).
// ===========================================================================
template <int CBLOCKS>
__global__ __launch_bounds__(128)
void conv3x3_wmma_kernel(const unsigned short* __restrict__ X,  // [B*HW][CpadIn] bf16
                         const unsigned int*  __restrict__ Wp,  // packed weights
                         const float*         __restrict__ bias,
                         unsigned short*      __restrict__ Y,   // [B*HW][CoutPad] bf16
                         int Hs, int Ws, int CoutPad, int CoutReal, int relu) {
  constexpr int CpadIn = CBLOCKS * 32;
  constexpr int nK = 9 * CBLOCKS;
  constexpr int LP = CpadIn + 8;   // padded channel pitch (bank skew; *2B multiple of 16)
  constexpr int HPX = 6 * 18;      // halo pixels
  constexpr int UNITS = CpadIn / 8;  // 16B units per pixel row
  __shared__ __align__(16) unsigned short smem[HPX * LP];

  const int HW = Hs * Ws;
  const int tid = threadIdx.x;
  const int lane = tid & 31;
  const int ntile = tid >> 5;
  const int xt = blockIdx.x * 16;
  const int yt = blockIdx.y * 4;
  const int b = blockIdx.z;
  const size_t ibase = (size_t)b * HW;

  // ---- stage 18x6 halo (zero-padded) into LDS ----
  for (int t = tid; t < HPX * UNITS; t += 128) {
    const int px = t / UNITS, u = t % UNITS;
    const int ly = px / 18, lx = px % 18;
    const int gy = yt + ly - 1, gx = xt + lx - 1;
    Q4 val = Q4{0u, 0u, 0u, 0u};
    if ((unsigned)gy < (unsigned)Hs && (unsigned)gx < (unsigned)Ws) {
      val = *reinterpret_cast<const Q4*>(X + (ibase + (size_t)gy * Ws + gx) * CpadIn + u * 8);
    }
    *reinterpret_cast<Q4*>(smem + px * LP + u * 8) = val;
  }
  __syncthreads();
  if (ntile * 16 >= CoutPad) return;

  const int m = lane & 15;
  const int ksel = lane >> 4;

  v8f acc[4];
#pragma unroll
  for (int mt = 0; mt < 4; ++mt) acc[mt] = v8f{0.f, 0.f, 0.f, 0.f, 0.f, 0.f, 0.f, 0.f};

  const unsigned int* wbase = Wp + (size_t)ntile * nK * 32 * 8;

  int kc = 0;
#pragma unroll
  for (int tap = 0; tap < 9; ++tap) {
    const int dy = tap / 3 - 1, dx = tap % 3 - 1;
#pragma unroll
    for (int cb = 0; cb < CBLOCKS; ++cb, ++kc) {
      // weight fragment: one load, reused by 4 WMMAs
      Frag ub;
      const Q4* wp4 = reinterpret_cast<const Q4*>(wbase + ((size_t)kc * 32 + lane) * 8);
      ub.q[0] = wp4[0];
      ub.q[1] = wp4[1];
      if (kc + 1 < nK) {  // near-cache prefetch of next weight chunk
        __builtin_prefetch(wbase + ((size_t)(kc + 1) * 32 + lane) * 8, 0, 3);
      }
#pragma unroll
      for (int mt = 0; mt < 4; ++mt) {
        const int lly = mt + 1 + dy;        // 0..5
        const int llx = m + 1 + dx;         // 0..17
        const unsigned short* ap = smem + (lly * 18 + llx) * LP + cb * 32;
        Frag ua;
        ua.q[0] = *reinterpret_cast<const Q4*>(ap + ksel * 16);
        ua.q[1] = *reinterpret_cast<const Q4*>(ap + ksel * 16 + 8);
        acc[mt] = __builtin_amdgcn_wmma_f32_16x16x32_bf16(
            false, ua.v, false, ub.v, (short)0, acc[mt], false, false);
      }
    }
  }

  // Store: 32-bit C/D layout: lane l -> col n=l&15; VGPR r -> row m = (l>>4)*8 + r
  const int n = lane & 15;
  const int half = lane >> 4;
  const int co = ntile * 16 + n;
  const float bv = (co < CoutReal) ? bias[co] : 0.f;
#pragma unroll
  for (int mt = 0; mt < 4; ++mt) {
    const int gy = yt + mt;
#pragma unroll
    for (int r = 0; r < 8; ++r) {
      const int gx = xt + half * 8 + r;
      float v = acc[mt][r] + bv;
      if (relu) v = v > 0.f ? v : 0.f;
      Y[(ibase + (size_t)gy * Ws + gx) * CoutPad + co] = f2bf(v);
    }
  }
}

// ===========================================================================
// Elementwise kernels
// ===========================================================================
__global__ void resize_bilinear_kernel(const float* __restrict__ in, float* __restrict__ out,
                                       int planes, int Hi, int Wi, int Ho, int Wo, int clip01) {
  const int total = planes * Ho * Wo;
  int idx = blockIdx.x * blockDim.x + threadIdx.x;
  if (idx >= total) return;
  const int xo = idx % Wo;
  const int yo = (idx / Wo) % Ho;
  const int pl = idx / (Wo * Ho);
  const float sy = (float)Hi / (float)Ho, sx = (float)Wi / (float)Wo;
  float fy = (yo + 0.5f) * sy - 0.5f;
  float fx = (xo + 0.5f) * sx - 0.5f;
  fy = fminf(fmaxf(fy, 0.f), (float)(Hi - 1));
  fx = fminf(fmaxf(fx, 0.f), (float)(Wi - 1));
  int y0 = (int)fy, x0 = (int)fx;
  int y1 = y0 + 1 < Hi ? y0 + 1 : Hi - 1;
  int x1 = x0 + 1 < Wi ? x0 + 1 : Wi - 1;
  const float wy = fy - (float)y0, wx = fx - (float)x0;
  const float* ip = in + (size_t)pl * Hi * Wi;
  float v = (1.f - wy) * ((1.f - wx) * ip[y0 * Wi + x0] + wx * ip[y0 * Wi + x1]) +
            wy * ((1.f - wx) * ip[y1 * Wi + x0] + wx * ip[y1 * Wi + x1]);
  if (clip01) v = fminf(fmaxf(v, 0.f), 1.f);
  out[idx] = v;
}

__global__ void down_sparse_kernel(const float* __restrict__ DL, const float* __restrict__ ML,
                                   float* __restrict__ DLs, float* __restrict__ MLs,
                                   int H, int W, int k, int Bn) {
  const int Hs = H / k, Ws2 = W / k;
  const int total = Bn * Hs * Ws2;
  int idx = blockIdx.x * blockDim.x + threadIdx.x;
  if (idx >= total) return;
  const int xs = idx % Ws2;
  const int ys = (idx / Ws2) % Hs;
  const int b = idx / (Ws2 * Hs);
  if (k == 1) {
    const size_t g = (size_t)b * H * W + (size_t)ys * W + xs;
    MLs[idx] = ML[g] > 0.f ? 1.f : 0.f;
    DLs[idx] = DL[g];
    return;
  }
  float s = 0.f, c = 0.f;
  for (int dy = 0; dy < k; ++dy)
    for (int dx = 0; dx < k; ++dx) {
      const size_t g = ((size_t)b * H + (ys * k + dy)) * W + (xs * k + dx);
      const float mm = ML[g] > 0.f ? 1.f : 0.f;
      s += DL[g] * mm;
      c += mm;
    }
  MLs[idx] = c > 0.f ? 1.f : 0.f;
  DLs[idx] = c > 0.f ? s / (c + 1e-6f) : 0.f;
}

__global__ void build_x_kernel(const float* __restrict__ Is, const float* __restrict__ DLs,
                               const float* __restrict__ MLs, const float* __restrict__ Es,
                               unsigned short* __restrict__ X, int HW, int Bn) {
  const int total = Bn * HW;
  int idx = blockIdx.x * blockDim.x + threadIdx.x;
  if (idx >= total) return;
  const int b = idx / HW, p = idx % HW;
  unsigned short* row = X + (size_t)idx * 32;
  row[0] = f2bf(Is[((size_t)b * 3 + 0) * HW + p]);
  row[1] = f2bf(Is[((size_t)b * 3 + 1) * HW + p]);
  row[2] = f2bf(Is[((size_t)b * 3 + 2) * HW + p]);
  row[3] = f2bf(DLs[idx]);
  row[4] = f2bf(MLs[idx]);
  row[5] = f2bf(Es[idx]);
#pragma unroll
  for (int c = 6; c < 32; ++c) row[c] = 0;
}

__global__ void init_d_first_kernel(const float* __restrict__ Es, float* __restrict__ D, int n) {
  int idx = blockIdx.x * blockDim.x + threadIdx.x;
  if (idx >= n) return;
  D[idx] = fminf(fmaxf(Es[idx] * 10.f, 0.f), 10.f);
}

__global__ void init_d_blend_kernel(const float* __restrict__ Es, const float* __restrict__ up,
                                    float* __restrict__ D, int Hs, int Ws2, int Bn) {
  const int HW = Hs * Ws2;
  const int total = Bn * HW;
  int idx = blockIdx.x * blockDim.x + threadIdx.x;
  if (idx >= total) return;
  const int p = idx % HW;
  const int y = p / Ws2, x = p % Ws2;
  const float e = Es[idx];
  const float gx = (x > 0) ? e - Es[idx - 1] : 0.f;
  const float gy = (y > 0) ? e - Es[idx - Ws2] : 0.f;
  const float g = fminf(fmaxf((fabsf(gx) + fabsf(gy)) * 0.5f, 0.f), 1.f);
  const float w = 0.7f * fminf(fmaxf(1.f - g * 10.f, 0.f), 1.f);  // EDGE_TAU=0.1, gamma=1
  const float Ps = fminf(fmaxf(e * 10.f, 0.f), 10.f);
  D[idx] = w * up[idx] + (1.f - w) * Ps;
}

// P row layout per pixel (36 f32): Anorm[24], center[3], sigma[3], kappa[3], pad[3]
__global__ void finalize_params_kernel(const unsigned short* __restrict__ aff,   // [pix][32]
                                       const unsigned short* __restrict__ gate,  // [pix][16]
                                       const unsigned short* __restrict__ curv,  // [pix][16]
                                       const float* __restrict__ D,
                                       const float* __restrict__ curv_w,         // (3,65,3,3)
                                       float* __restrict__ P, int Hs, int Ws2, int Bn) {
  const int HW = Hs * Ws2;
  const int total = Bn * HW;
  int idx = blockIdx.x * blockDim.x + threadIdx.x;
  if (idx >= total) return;
  const unsigned short* ar = aff + (size_t)idx * 32;
  float* pr = P + (size_t)idx * 36;
  float A[24];
#pragma unroll
  for (int j = 0; j < 24; ++j) A[j] = bf2f(ar[j]);
#pragma unroll
  for (int k = 0; k < 3; ++k) {
    float s = 0.f;
#pragma unroll
    for (int j = 0; j < 8; ++j) s += fabsf(A[k * 8 + j]);
    const float inv = 1.f / (s + 1e-6f);
    float csum = 0.f;
#pragma unroll
    for (int j = 0; j < 8; ++j) {
      const float an = A[k * 8 + j] * inv;
      pr[k * 8 + j] = an;
      csum += an;
    }
    pr[24 + k] = 1.f - csum;
  }
  // gate softmax
  float g0 = bf2f(gate[(size_t)idx * 16 + 0]);
  float g1 = bf2f(gate[(size_t)idx * 16 + 1]);
  float g2 = bf2f(gate[(size_t)idx * 16 + 2]);
  const float gm = fmaxf(g0, fmaxf(g1, g2));
  const float e0 = expf(g0 - gm), e1 = expf(g1 - gm), e2 = expf(g2 - gm);
  const float es = e0 + e1 + e2;
  pr[27] = e0 / es; pr[28] = e1 / es; pr[29] = e2 / es;
  // kappa: curv conv over feat (done) + scalar 3x3 conv over D/DMAX (channel 64)
  const int b = idx / HW, p = idx % HW;
  const int y = p / Ws2, x = p % Ws2;
#pragma unroll
  for (int k = 0; k < 3; ++k) {
    float r = bf2f(curv[(size_t)idx * 16 + k]);
    for (int tap = 0; tap < 9; ++tap) {
      const int ny = y + tap / 3 - 1, nx = x + tap % 3 - 1;
      if ((unsigned)ny < (unsigned)Hs && (unsigned)nx < (unsigned)Ws2) {
        r += curv_w[((size_t)k * 65 + 64) * 9 + tap] *
             (D[(size_t)b * HW + ny * Ws2 + nx] * 0.1f);
      }
    }
    pr[30 + k] = 0.1f + 0.9f * (1.f / (1.f + expf(-r)));
  }
  pr[33] = 0.f; pr[34] = 0.f; pr[35] = 0.f;
}

__global__ void prop_step_kernel(const float* __restrict__ Din, const float* __restrict__ P,
                                 const float* __restrict__ DLs, const float* __restrict__ MLs,
                                 float* __restrict__ Dout, int Hs, int Ws2, int Bn) {
  const int HW = Hs * Ws2;
  const int total = Bn * HW;
  int idx = blockIdx.x * blockDim.x + threadIdx.x;
  if (idx >= total) return;
  const int b = idx / HW, p = idx % HW;
  const int y = p / Ws2, x = p % Ws2;
  const float* pr = P + (size_t)idx * 36;
  const float Dc = Din[idx];
  const int offy[8] = {-1, -1, -1, 0, 0, 1, 1, 1};
  const int offx[8] = {-1, 0, 1, -1, 1, -1, 0, 1};
  const int dils[3] = {1, 2, 4};
  float mix = 0.f;
#pragma unroll
  for (int k = 0; k < 3; ++k) {
    const int d = dils[k];
    float agg = pr[24 + k] * Dc;
#pragma unroll
    for (int j = 0; j < 8; ++j) {
      const int ny = y + offy[j] * d, nx = x + offx[j] * d;
      float v = 0.f;
      if ((unsigned)ny < (unsigned)Hs && (unsigned)nx < (unsigned)Ws2)
        v = Din[(size_t)b * HW + ny * Ws2 + nx];
      agg += pr[k * 8 + j] * v;
    }
    mix += pr[27 + k] * (Dc + pr[30 + k] * (agg - Dc));
  }
  const float ml = MLs[idx];
  Dout[idx] = (ml > 0.f) ? (0.9f * DLs[idx] + 0.1f * mix) : mix;
}

// ===========================================================================
// Host orchestration
// ===========================================================================
static inline int cdiv(int a, int b) { return (a + b - 1) / b; }

extern "C" void kernel_launch(void* const* d_in, const int* in_sizes, int n_in,
                              void* d_out, int out_size, void* d_ws, size_t ws_size,
                              hipStream_t stream) {
  (void)in_sizes; (void)n_in; (void)out_size; (void)ws_size;
  const int Bn = 2, H = 480, W = 640;
  const int NMAX = Bn * H * W;  // 614400

  const float* I      = (const float*)d_in[0];
  const float* DL     = (const float*)d_in[1];
  const float* ML     = (const float*)d_in[2];
  const float* Enorm  = (const float*)d_in[3];
  const float* enc_w0 = (const float*)d_in[4];
  const float* enc_b0 = (const float*)d_in[5];
  const float* enc_w1 = (const float*)d_in[6];
  const float* enc_b1 = (const float*)d_in[7];
  const float* enc_w2 = (const float*)d_in[8];
  const float* enc_b2 = (const float*)d_in[9];
  const float* aff_w  = (const float*)d_in[10];
  const float* aff_b  = (const float*)d_in[11];
  const float* gate_w = (const float*)d_in[12];
  const float* gate_b = (const float*)d_in[13];
  const float* curv_w = (const float*)d_in[14];
  const float* curv_b = (const float*)d_in[15];

  // ---- scratch arena -------------------------------------------------------
  char* base = (char*)d_ws;
  size_t off = 0;
  auto alloc = [&](size_t bytes) -> char* {
    off = (off + 255) & ~(size_t)255;
    char* p = base + off;
    off += bytes;
    return p;
  };
  // packed weights: dwords = ntiles*nK*32*8
  unsigned int* wp_enc0 = (unsigned int*)alloc((size_t)4 * 9  * 32 * 8 * 4);
  unsigned int* wp_enc1 = (unsigned int*)alloc((size_t)4 * 18 * 32 * 8 * 4);
  unsigned int* wp_enc2 = (unsigned int*)alloc((size_t)4 * 18 * 32 * 8 * 4);
  unsigned int* wp_aff  = (unsigned int*)alloc((size_t)2 * 18 * 32 * 8 * 4);
  unsigned int* wp_gate = (unsigned int*)alloc((size_t)1 * 18 * 32 * 8 * 4);
  unsigned int* wp_curv = (unsigned int*)alloc((size_t)1 * 18 * 32 * 8 * 4);
  unsigned short* X32      = (unsigned short*)alloc((size_t)NMAX * 32 * 2);
  unsigned short* Fa       = (unsigned short*)alloc((size_t)NMAX * 64 * 2);
  unsigned short* Fb       = (unsigned short*)alloc((size_t)NMAX * 64 * 2);
  unsigned short* aff_raw  = (unsigned short*)alloc((size_t)NMAX * 32 * 2);
  unsigned short* gate_raw = (unsigned short*)alloc((size_t)NMAX * 16 * 2);
  unsigned short* curv_raw = (unsigned short*)alloc((size_t)NMAX * 16 * 2);
  float* Pbuf = (float*)alloc((size_t)NMAX * 36 * 4);
  float* Ibuf = (float*)alloc((size_t)NMAX * 3 * 4);
  float* Ebuf = (float*)alloc((size_t)NMAX * 4);
  float* DLsb = (float*)alloc((size_t)NMAX * 4);
  float* MLsb = (float*)alloc((size_t)NMAX * 4);
  float* D0   = (float*)alloc((size_t)NMAX * 4);
  float* D1   = (float*)alloc((size_t)NMAX * 4);
  float* Dup  = (float*)alloc((size_t)NMAX * 4);

  const int T = 256;
  // ---- pack all weights once ----------------------------------------------
  {
    int n;
    n = 4 * 9 * 32;
    pack_weights_kernel<<<cdiv(n, T), T, 0, stream>>>(enc_w0, wp_enc0, 6, 6, 32, 64, 4);
    n = 4 * 18 * 32;
    pack_weights_kernel<<<cdiv(n, T), T, 0, stream>>>(enc_w1, wp_enc1, 64, 64, 64, 64, 4);
    pack_weights_kernel<<<cdiv(n, T), T, 0, stream>>>(enc_w2, wp_enc2, 64, 64, 64, 64, 4);
    n = 2 * 18 * 32;
    pack_weights_kernel<<<cdiv(n, T), T, 0, stream>>>(aff_w, wp_aff, 64, 64, 64, 24, 2);
    n = 1 * 18 * 32;
    pack_weights_kernel<<<cdiv(n, T), T, 0, stream>>>(gate_w, wp_gate, 64, 64, 64, 3, 1);
    // curv: only first 64 input channels go through WMMA (CinLog=65)
    pack_weights_kernel<<<cdiv(n, T), T, 0, stream>>>(curv_w, wp_curv, 65, 64, 64, 3, 1);
  }

  const int scales[3] = {4, 2, 1};
  int Hp = 0, Wpv = 0;
  for (int si = 0; si < 3; ++si) {
    const int s = scales[si];
    const int Hs = H / s, Ws2 = W / s;
    const int HW = Hs * Ws2;
    const int Npix = Bn * HW;

    // resample inputs to this scale
    resize_bilinear_kernel<<<cdiv(Bn * 3 * HW, T), T, 0, stream>>>(I, Ibuf, Bn * 3, H, W, Hs, Ws2, 0);
    resize_bilinear_kernel<<<cdiv(Npix, T), T, 0, stream>>>(Enorm, Ebuf, Bn, H, W, Hs, Ws2, 1);
    down_sparse_kernel<<<cdiv(Npix, T), T, 0, stream>>>(DL, ML, DLsb, MLsb, H, W, s, Bn);

    // D init
    if (si == 0) {
      init_d_first_kernel<<<cdiv(Npix, T), T, 0, stream>>>(Ebuf, D0, Npix);
    } else {
      resize_bilinear_kernel<<<cdiv(Npix, T), T, 0, stream>>>(D0, Dup, Bn, Hp, Wpv, Hs, Ws2, 0);
      init_d_blend_kernel<<<cdiv(Npix, T), T, 0, stream>>>(Ebuf, Dup, D0, Hs, Ws2, Bn);
    }

    // encoder + heads (WMMA path); 16x4 pixel tiles per block
    build_x_kernel<<<cdiv(Npix, T), T, 0, stream>>>(Ibuf, DLsb, MLsb, Ebuf, X32, HW, Bn);
    dim3 cgrid(Ws2 / 16, Hs / 4, Bn);
    conv3x3_wmma_kernel<1><<<cgrid, 128, 0, stream>>>(X32, wp_enc0, enc_b0, Fa, Hs, Ws2, 64, 64, 1);
    conv3x3_wmma_kernel<2><<<cgrid, 128, 0, stream>>>(Fa,  wp_enc1, enc_b1, Fb, Hs, Ws2, 64, 64, 1);
    conv3x3_wmma_kernel<2><<<cgrid, 128, 0, stream>>>(Fb,  wp_enc2, enc_b2, Fa, Hs, Ws2, 64, 64, 1);
    conv3x3_wmma_kernel<2><<<cgrid, 128, 0, stream>>>(Fa,  wp_aff,  aff_b,  aff_raw,  Hs, Ws2, 32, 24, 0);
    conv3x3_wmma_kernel<2><<<cgrid, 128, 0, stream>>>(Fa,  wp_gate, gate_b, gate_raw, Hs, Ws2, 16, 3, 0);
    conv3x3_wmma_kernel<2><<<cgrid, 128, 0, stream>>>(Fa,  wp_curv, curv_b, curv_raw, Hs, Ws2, 16, 3, 0);

    finalize_params_kernel<<<cdiv(Npix, T), T, 0, stream>>>(aff_raw, gate_raw, curv_raw,
                                                            D0, curv_w, Pbuf, Hs, Ws2, Bn);
    // 6 propagation steps, ping-pong D0/D1; result ends in D0
    for (int step = 0; step < 6; ++step) {
      const float* din = (step % 2 == 0) ? D0 : D1;
      float* dout = (step % 2 == 0) ? D1 : D0;
      if (si == 2 && step == 5) dout = (float*)d_out;
      prop_step_kernel<<<cdiv(Npix, T), T, 0, stream>>>(din, Pbuf, DLsb, MLsb, dout, Hs, Ws2, Bn);
    }
    Hp = Hs; Wpv = Ws2;
  }
}
